// Block_26001732010180
// MI455X (gfx1250) — matrix operand order repaired and checked by reading, not compiled, not run on hardware
//
#include <hip/hip_runtime.h>
#include <hip/hip_bf16.h>
#include <cstdint>
#include <cstddef>

// ---------------------------------------------------------------------------
// Types / helpers for CDNA5 WMMA (wave32, 16x16x32 bf16 -> f32)
// ---------------------------------------------------------------------------
typedef __bf16 bf16;
typedef __attribute__((ext_vector_type(16))) __bf16 bf16x16;
typedef __attribute__((ext_vector_type(8)))  __bf16 bf16x8;
typedef __attribute__((ext_vector_type(8)))  float  f32x8;
typedef __attribute__((ext_vector_type(4)))  int    i32x4;

// address-space-qualified v4i for the async-to-LDS builtin (sig from clang
// diagnostic: (v4i addrspace(1)*, v4i addrspace(3)*, imm offset, imm cpol))
typedef __attribute__((address_space(1))) i32x4 g_i32x4;
typedef __attribute__((address_space(3))) i32x4 l_i32x4;

#if __has_builtin(__builtin_amdgcn_global_load_async_to_lds_b128) && \
    __has_builtin(__builtin_amdgcn_s_wait_asynccnt)
#define USE_ASYNC_LDS 1
#else
#define USE_ASYNC_LDS 0
#endif

__device__ __forceinline__ bf16 f2bf(float f) {
  unsigned u = __float_as_uint(f);
  unsigned r = (u + 0x7FFFu + ((u >> 16) & 1u)) >> 16;   // RNE
  unsigned short s = (unsigned short)r;
  return __builtin_bit_cast(bf16, s);
}
__device__ __forceinline__ float bf2f(bf16 h) {
  unsigned short s = __builtin_bit_cast(unsigned short, h);
  return __uint_as_float(((unsigned)s) << 16);
}
__device__ __forceinline__ bf16 bf0() {
  return __builtin_bit_cast(bf16, (unsigned short)0);
}

#define WMMA_BF16(a, b, c) \
  __builtin_amdgcn_wmma_f32_16x16x32_bf16(false, (a), false, (b), (short)0, (c), false, false)

// A fragment (16x32, row-major tile in LDS, leading dim ld):
// lane l: M = l%16; halves 0..7 = K[(l/16)*8 .. +7], halves 8..15 = K[16+(l/16)*8 .. +7]
__device__ __forceinline__ bf16x16 load_frag_A(const bf16* smem, int ld, int mBase,
                                               int kBase, int lane) {
  const int m  = mBase + (lane & 15);
  const int k0 = kBase + ((lane >> 4) << 3);
  const bf16* p = smem + (size_t)m * ld + k0;
  bf16x8 lo = *(const bf16x8*)p;
  bf16x8 hi = *(const bf16x8*)(p + 16);
  bf16x16 r;
#pragma unroll
  for (int i = 0; i < 8; ++i) { r[i] = lo[i]; r[i + 8] = hi[i]; }
  return r;
}

// B fragment (32x16) from a K-transposed LDS tile Bt[n][k] (leading dim ld):
// lane l: column n = l%16; halves 0..7 = K[(l/16)*8 .. +7], halves 8..15 = K[16+(l/16)*8 .. +7]
__device__ __forceinline__ bf16x16 load_frag_B(const bf16* smem, int ld, int nBase,
                                               int kBase, int lane) {
  const bf16* p = smem + (size_t)(nBase + (lane & 15)) * ld + kBase + ((lane >> 4) << 3);
  bf16x8 lo = *(const bf16x8*)p;
  bf16x8 hi = *(const bf16x8*)(p + 16);
  bf16x16 r;
#pragma unroll
  for (int i = 0; i < 8; ++i) { r[i] = lo[i]; r[i + 8] = hi[i]; }
  return r;
}

// ---------------------------------------------------------------------------
// Stage a 128 x 64 bf16 tile (rows contiguous, global row stride ldg elems)
// into LDS. 256 threads, 4 x b128 per thread. Rows >= validRows are zeroed.
// Uses GLOBAL_LOAD_ASYNC_TO_LDS_B128 when the toolchain exposes it.
// ---------------------------------------------------------------------------
__device__ __forceinline__ void stage_tile64(bf16* __restrict__ dst,
                                             const bf16* __restrict__ src,
                                             int ldg, int validRows, int tid) {
#pragma unroll
  for (int i = 0; i < 4; ++i) {
    const int chunk = i * 256 + tid;          // 0..1023
    const int row = chunk >> 3;               // 0..127
    const int seg = (chunk & 7) << 3;         // element offset 0..56
    bf16* d = dst + row * 64 + seg;
    if (row < validRows) {
      const bf16* s = src + (size_t)row * ldg + seg;
      __builtin_prefetch(s + 64, 0, 1);       // next K-tile -> global_prefetch_b8
#if USE_ASYNC_LDS
      __builtin_amdgcn_global_load_async_to_lds_b128((g_i32x4*)s, (l_i32x4*)d, 0, 0);
#else
      *(bf16x8*)d = *(const bf16x8*)s;
#endif
    } else {
      const bf16x8 z = {};
      *(bf16x8*)d = z;
    }
  }
}

__device__ __forceinline__ void stage_fence() {
#if USE_ASYNC_LDS
  __builtin_amdgcn_s_wait_asynccnt(0);
#endif
}

// ---------------------------------------------------------------------------
// Tiled WMMA GEMM:  C[M,N] = act(A[M,K] @ W + bias + res),  W given PRE-
// TRANSPOSED as Bt[N][K] bf16 so both LDS stages are contiguous row copies.
// 256 threads = 8 waves; block tile 128x128, BK=64; wave tile 32x64.
// ---------------------------------------------------------------------------
template <bool BIAS, bool RES, bool GELU_, bool OUTBF>
__global__ __launch_bounds__(256) void gemm_bf16_wmma(
    const bf16* __restrict__ A, const bf16* __restrict__ Bt,
    const float* __restrict__ bias, const float* __restrict__ res,
    float* __restrict__ Cf, bf16* __restrict__ Cb, int M, int N, int K) {
  constexpr int BM = 128, BN = 128, BK = 64;
  __shared__ alignas(16) bf16 As[BM * BK];
  __shared__ alignas(16) bf16 Bs[BN * BK];
  const int tid = threadIdx.x, lane = tid & 31, wid = tid >> 5;
  const int wr = wid >> 1;   // 0..3 : wave row (32 rows each)
  const int wc = wid & 1;    // 0..1 : wave col (64 cols each)
  const int m0 = blockIdx.y * BM, n0 = blockIdx.x * BN;
  const int aRows = (M - m0) < BM ? (M - m0) : BM;
  const int bRows = (N - n0) < BN ? (N - n0) : BN;

  const f32x8 zacc = {};
  f32x8 acc[2][4];
#pragma unroll
  for (int i = 0; i < 2; ++i)
#pragma unroll
    for (int j = 0; j < 4; ++j) acc[i][j] = zacc;

  for (int kt = 0; kt < K; kt += BK) {
    stage_tile64(As, A + (size_t)m0 * K + kt, K, aRows, tid);
    stage_tile64(Bs, Bt + (size_t)n0 * K + kt, K, bRows, tid);
    stage_fence();
    __syncthreads();
#pragma unroll
    for (int kk = 0; kk < BK; kk += 32) {
      bf16x16 af[2], bfr[4];
#pragma unroll
      for (int mt = 0; mt < 2; ++mt)
        af[mt] = load_frag_A(As, BK, wr * 32 + mt * 16, kk, lane);
#pragma unroll
      for (int nt = 0; nt < 4; ++nt)
        bfr[nt] = load_frag_B(Bs, BK, wc * 64 + nt * 16, kk, lane);
#pragma unroll
      for (int mt = 0; mt < 2; ++mt)
#pragma unroll
        for (int nt = 0; nt < 4; ++nt)
          acc[mt][nt] = WMMA_BF16(af[mt], bfr[nt], acc[mt][nt]);
    }
    __syncthreads();
  }

  // Epilogue. C/D layout: VGPR r holds M = r + 8*(lane/16), N = lane%16.
  const int mh = (lane >> 4) << 3;
  const int nl = lane & 15;
#pragma unroll
  for (int mt = 0; mt < 2; ++mt)
#pragma unroll
    for (int nt = 0; nt < 4; ++nt) {
      const int col = n0 + wc * 64 + nt * 16 + nl;
      if (col >= N) continue;
      const float bv = BIAS ? bias[col] : 0.f;
#pragma unroll
      for (int r = 0; r < 8; ++r) {
        const int row = m0 + wr * 32 + mt * 16 + mh + r;
        if (row >= M) continue;
        float v = acc[mt][nt][r] + bv;
        if (RES) v += res[(size_t)row * N + col];
        if (GELU_) v = 0.5f * v * (1.f + erff(v * 0.70710678118654752f));
        const size_t o = (size_t)row * N + col;
        if (OUTBF) Cb[o] = f2bf(v); else Cf[o] = v;
      }
    }
}

// ---------------------------------------------------------------------------
// Weight transpose + fp32->bf16:  in[K][N] fp32  ->  out[N][K] bf16.
// LDS-tiled 32x32 so both global reads and writes are coalesced.
// ---------------------------------------------------------------------------
__global__ __launch_bounds__(256) void transpose_cvt_kernel(
    const float* __restrict__ in, bf16* __restrict__ out, int K, int N) {
  __shared__ float tile[32][33];
  const int k0 = blockIdx.y * 32, n0 = blockIdx.x * 32;
  const int tx = threadIdx.x, ty = threadIdx.y;   // blockDim (32, 8)
  for (int i = ty; i < 32; i += 8) {
    const int k = k0 + i, n = n0 + tx;
    tile[i][tx] = (k < K && n < N) ? in[(size_t)k * N + n] : 0.f;
  }
  __syncthreads();
  for (int i = ty; i < 32; i += 8) {
    const int n = n0 + i, k = k0 + tx;
    if (n < N && k < K) out[(size_t)n * K + k] = f2bf(tile[tx][i]);
  }
}

// ---------------------------------------------------------------------------
// LayerNorm over rows of 768 (256 threads, 3 elems/thread), wave32 shuffles.
// ---------------------------------------------------------------------------
__device__ __forceinline__ void ln_row_body(const float* __restrict__ x,
                                            const float* __restrict__ g,
                                            const float* __restrict__ bb,
                                            float* __restrict__ outF,
                                            bf16* __restrict__ outB, float* red) {
  const int tid = threadIdx.x;
  const float v0 = x[tid], v1 = x[tid + 256], v2 = x[tid + 512];
  float s = v0 + v1 + v2;
#pragma unroll
  for (int o = 16; o > 0; o >>= 1) s += __shfl_xor(s, o, 32);
  if ((tid & 31) == 0) red[tid >> 5] = s;
  __syncthreads();
  if (tid == 0) {
    float t = 0.f;
    for (int i = 0; i < 8; ++i) t += red[i];
    red[8] = t * (1.f / 768.f);
  }
  __syncthreads();
  const float mu = red[8];
  const float d0 = v0 - mu, d1 = v1 - mu, d2 = v2 - mu;
  float s2 = d0 * d0 + d1 * d1 + d2 * d2;
#pragma unroll
  for (int o = 16; o > 0; o >>= 1) s2 += __shfl_xor(s2, o, 32);
  if ((tid & 31) == 0) red[tid >> 5] = s2;
  __syncthreads();
  if (tid == 0) {
    float t = 0.f;
    for (int i = 0; i < 8; ++i) t += red[i];
    red[9] = rsqrtf(t * (1.f / 768.f) + 1e-5f);
  }
  __syncthreads();
  const float rs = red[9];
  const float o0 = d0 * rs * g[tid] + bb[tid];
  const float o1 = d1 * rs * g[tid + 256] + bb[tid + 256];
  const float o2 = d2 * rs * g[tid + 512] + bb[tid + 512];
  if (outF) { outF[tid] = o0; outF[tid + 256] = o1; outF[tid + 512] = o2; }
  if (outB) { outB[tid] = f2bf(o0); outB[tid + 256] = f2bf(o1); outB[tid + 512] = f2bf(o2); }
}

__global__ __launch_bounds__(256) void ln_rows_kernel(
    const float* __restrict__ in, size_t rowStride, const float* __restrict__ g,
    const float* __restrict__ b, float* __restrict__ outF, bf16* __restrict__ outB,
    int rows) {
  __shared__ float red[10];
  const int r = blockIdx.x;
  if (r >= rows) return;
  ln_row_body(in + (size_t)r * rowStride, g, b,
              outF ? outF + (size_t)r * 768 : nullptr,
              outB ? outB + (size_t)r * 768 : nullptr, red);
}

// LN of the spatial-attention input xs (gathered from cls token / xt_full).
__global__ __launch_bounds__(256) void ln_xs_kernel(
    const float* __restrict__ x, const float* __restrict__ xt_full,
    const float* __restrict__ g, const float* __restrict__ b,
    bf16* __restrict__ outB) {
  __shared__ float red[10];
  const int row = blockIdx.x;             // 0..12607  (inst*197 + p)
  const int inst = row / 197, p = row % 197;
  const int bb = inst >> 4, t = inst & 15;
  const float* src = (p == 0)
      ? (x + (size_t)bb * 3137 * 768)
      : (xt_full + ((size_t)bb * 3136 + (size_t)(p - 1) * 16 + t) * 768);
  ln_row_body(src, g, b, nullptr, outB + (size_t)row * 768, red);
}

// ---------------------------------------------------------------------------
// Elementwise / gather kernels
// ---------------------------------------------------------------------------
// x[:,1:,:] -> contiguous (12544, 768) fp32
__global__ void strip_cls_kernel(const float* __restrict__ x, float* __restrict__ out) {
  const size_t n = (size_t)12544 * 768;
  for (size_t i = (size_t)blockIdx.x * 256 + threadIdx.x; i < n;
       i += (size_t)gridDim.x * 256) {
    const size_t r = i / 768, c = i % 768;
    const size_t b = r / 3136, rem = r % 3136;
    out[i] = x[((size_t)b * 3137 + 1 + rem) * 768 + c];
  }
}

// gather last-L timesteps of each temporal sequence from xtn (bf16)
__global__ void gather_win_kernel(const bf16* __restrict__ xtn, bf16* __restrict__ out,
                                  int L) {
  const size_t n = (size_t)784 * L * 768;
  for (size_t i = (size_t)blockIdx.x * 256 + threadIdx.x; i < n;
       i += (size_t)gridDim.x * 256) {
    const size_t r = i / 768, c = i % 768;
    const size_t seq = r / L;
    const int t = 16 - L + (int)(r % L);
    out[i] = xtn[((size_t)seq * 16 + t) * 768 + c];
  }
}

// hierarchical temporal merge -> bf16 res_temp
__global__ void merge_temporal_kernel(const float* __restrict__ x16f,
                                      const float* __restrict__ x8f,
                                      const float* __restrict__ x4f,
                                      bf16* __restrict__ out) {
  const size_t n = (size_t)12544 * 768;
  for (size_t i = (size_t)blockIdx.x * 256 + threadIdx.x; i < n;
       i += (size_t)gridDim.x * 256) {
    const size_t r = i / 768, c = i % 768;
    const int t = (int)(r & 15);
    const size_t seq = r >> 4;
    float v = x16f[i];
    if (t >= 8) {
      const float v8 = x8f[((size_t)seq * 8 + (t - 8)) * 768 + c];
      if (t >= 12) {
        const float v4 = x4f[((size_t)seq * 4 + (t - 12)) * 768 + c];
        v = 0.5f * v + 0.25f * v8 + 0.25f * v4;
      } else {
        v = 0.5f * v + 0.5f * v8;
      }
    }
    out[i] = f2bf(v);
  }
}

// x_cat = concat[init_cls, xt_full] + concat[cls_out, x_s(permuted res_spatial)]
__global__ void build_xcat_kernel(const float* __restrict__ x,
                                  const float* __restrict__ xt_full,
                                  const float* __restrict__ res_sp,
                                  const float* __restrict__ cls_out,
                                  float* __restrict__ xcat) {
  const size_t n = (size_t)12548 * 768;
  for (size_t i = (size_t)blockIdx.x * 256 + threadIdx.x; i < n;
       i += (size_t)gridDim.x * 256) {
    const size_t row = i / 768, c = i % 768;
    const size_t b = row / 3137;
    const int p = (int)(row % 3137);
    float v;
    if (p == 0) {
      v = x[i] + cls_out[b * 768 + c];
    } else {
      const int idx = p - 1;
      const int k = idx >> 4, t = idx & 15;
      v = xt_full[((size_t)b * 3136 + idx) * 768 + c] +
          res_sp[(((size_t)(b * 16 + t)) * 197 + 1 + k) * 768 + c];
    }
    xcat[i] = v;
  }
}

// ---------------------------------------------------------------------------
// Temporal attention (L <= 16): tiny, VALU + LDS. qkv bf16 (rows x 2304).
// grid = (784, 12), block = 128.
// ---------------------------------------------------------------------------
__global__ __launch_bounds__(128) void temporal_attn_kernel(
    const bf16* __restrict__ qkv, bf16* __restrict__ out, int L, float scale) {
  const int seq = blockIdx.x, h = blockIdx.y;
  const int tid = threadIdx.x;
  __shared__ float q[16 * 64], k[16 * 64], v[16 * 64], S[16 * 16];
  const int cb = h * 64;
  for (int i = tid; i < L * 64; i += 128) {
    const int t = i >> 6, d = i & 63;
    const bf16* row = qkv + ((size_t)seq * L + t) * 2304;
    q[i] = bf2f(row[cb + d]);
    k[i] = bf2f(row[768 + cb + d]);
    v[i] = bf2f(row[1536 + cb + d]);
  }
  __syncthreads();
  for (int i = tid; i < L * L; i += 128) {
    const int t = i / L, u = i % L;
    float s = 0.f;
#pragma unroll
    for (int d = 0; d < 64; ++d) s += q[t * 64 + d] * k[u * 64 + d];
    S[t * 16 + u] = s * scale;
  }
  __syncthreads();
  if (tid < L) {
    float mx = -3.0e38f;
    for (int u = 0; u < L; ++u) mx = fmaxf(mx, S[tid * 16 + u]);
    float sum = 0.f;
    for (int u = 0; u < L; ++u) {
      const float e = __expf(S[tid * 16 + u] - mx);
      S[tid * 16 + u] = e;
      sum += e;
    }
    const float inv = 1.f / sum;
    for (int u = 0; u < L; ++u) S[tid * 16 + u] *= inv;
  }
  __syncthreads();
  for (int i = tid; i < L * 64; i += 128) {
    const int t = i >> 6, d = i & 63;
    float o = 0.f;
    for (int u = 0; u < L; ++u) o += S[t * 16 + u] * v[u * 64 + d];
    out[((size_t)seq * L + t) * 768 + cb + d] = f2bf(o);
  }
}

// ---------------------------------------------------------------------------
// Spatial attention (197 keys) with WMMA. grid = (13 qtiles, 12 heads, 64 inst),
// block = 128 (4 waves). Two-pass: S = Q K^T (wmma) -> softmax -> O = P V (wmma).
// ---------------------------------------------------------------------------
__global__ __launch_bounds__(128) void spatial_attn_kernel(
    const bf16* __restrict__ qkv, bf16* __restrict__ out, float scale) {
  constexpr int SP = 224;  // 197 keys padded to 224 (7 * 32)
  const int qt = blockIdx.x, h = blockIdx.y, inst = blockIdx.z;
  const int tid = threadIdx.x, lane = tid & 31, wid = tid >> 5;
  __shared__ alignas(16) bf16 Qs[16 * 64];
  __shared__ alignas(16) bf16 KV[4][16 * 64];
  __shared__ alignas(16) float S[16 * SP];
  __shared__ alignas(16) bf16 P[16 * SP];
  const size_t rowBase = (size_t)inst * 197;
  const int cb = h * 64;

  for (int i = tid; i < 16 * 64; i += 128) {
    const int r = i >> 6, d = i & 63;
    const int gq = qt * 16 + r;
    Qs[i] = (gq < 197) ? qkv[(rowBase + gq) * 2304 + cb + d] : bf0();
  }
  __syncthreads();

  // pass A: score tiles, waves strided over 13 key tiles
  for (int jt = wid; jt < 13; jt += 4) {
    for (int i = lane; i < 16 * 64; i += 32) {
      const int r = i >> 6, d = i & 63;
      const int gk = jt * 16 + r;
      KV[wid][i] = (gk < 197) ? qkv[(rowBase + gk) * 2304 + 768 + cb + d] : bf0();
    }
    f32x8 c = {};
    bf16x16 a0 = load_frag_A(Qs, 64, 0, 0, lane);
    bf16x16 a1 = load_frag_A(Qs, 64, 0, 32, lane);
    bf16x16 b0 = load_frag_B(KV[wid], 64, 0, 0, lane);
    bf16x16 b1 = load_frag_B(KV[wid], 64, 0, 32, lane);
    c = WMMA_BF16(a0, b0, c);
    c = WMMA_BF16(a1, b1, c);
    const int mh = (lane >> 4) << 3, nl = lane & 15;
    const int gk = jt * 16 + nl;
#pragma unroll
    for (int r = 0; r < 8; ++r)
      S[(mh + r) * SP + jt * 16 + nl] = (gk < 197) ? c[r] * scale : -3.0e38f;
  }
  for (int i = tid; i < 16 * 16; i += 128) {  // pad columns 208..223
    const int m = i >> 4, nn = 208 + (i & 15);
    S[m * SP + nn] = -3.0e38f;
  }
  __syncthreads();

  if (tid < 16) {  // row-wise softmax
    float mx = -3.0e38f;
    for (int j = 0; j < SP; ++j) mx = fmaxf(mx, S[tid * SP + j]);
    float sum = 0.f;
    for (int j = 0; j < SP; ++j) {
      const float e = __expf(S[tid * SP + j] - mx);
      S[tid * SP + j] = e;
      sum += e;
    }
    const float inv = 1.f / sum;
    for (int j = 0; j < SP; ++j) P[tid * SP + j] = f2bf(S[tid * SP + j] * inv);
  }
  __syncthreads();

  // pass B: O = P V; wave wid owns dims [wid*16, wid*16+16)
  f32x8 oc = {};
  for (int kt = 0; kt < SP; kt += 32) {
    for (int i = lane; i < 16 * 32; i += 32) {  // stage V^T chunk: Vt[d][k]
      const int dl = i >> 5, kk = i & 31;
      const int gk = kt + kk;
      KV[wid][dl * 32 + kk] =
          (gk < 197) ? qkv[(rowBase + gk) * 2304 + 1536 + cb + wid * 16 + dl] : bf0();
    }
    bf16x16 a = load_frag_A(P, SP, 0, kt, lane);
    bf16x16 b = load_frag_B(KV[wid], 32, 0, 0, lane);
    oc = WMMA_BF16(a, b, oc);
  }
  const int mh = (lane >> 4) << 3, nl = lane & 15;
#pragma unroll
  for (int r = 0; r < 8; ++r) {
    const int gq = qt * 16 + mh + r;
    if (gq < 197) out[(rowBase + gq) * 768 + cb + wid * 16 + nl] = f2bf(oc[r]);
  }
}

// ---------------------------------------------------------------------------
// CLS token self-attention over T=16 (per batch, no scale). grid=4, block=256.
// ---------------------------------------------------------------------------
__global__ __launch_bounds__(256) void cls_attn_kernel(const float* __restrict__ cls_sp,
                                                       float* __restrict__ cls_out) {
  const int b = blockIdx.x;
  const int tid = threadIdx.x;
  __shared__ float sc[16];
  const int t = tid >> 4, l = tid & 15;
  const float* tgt = cls_sp + ((size_t)b * 16 + 15) * 768;
  const float* rowp = cls_sp + ((size_t)b * 16 + t) * 768;
  float p = 0.f;
  for (int c = l; c < 768; c += 16) p += tgt[c] * rowp[c];
#pragma unroll
  for (int o = 8; o > 0; o >>= 1) p += __shfl_xor(p, o, 16);
  if (l == 0) sc[t] = p;
  __syncthreads();
  if (tid == 0) {
    float mx = -3.0e38f;
    for (int i = 0; i < 16; ++i) mx = fmaxf(mx, sc[i]);
    float sum = 0.f;
    for (int i = 0; i < 16; ++i) { sc[i] = __expf(sc[i] - mx); sum += sc[i]; }
    const float inv = 1.f / sum;
    for (int i = 0; i < 16; ++i) sc[i] *= inv;
  }
  __syncthreads();
  for (int c = tid; c < 768; c += 256) {
    float o = 0.f;
    for (int i = 0; i < 16; ++i) o += sc[i] * cls_sp[((size_t)b * 16 + i) * 768 + c];
    cls_out[(size_t)b * 768 + c] = o;
  }
}

// ---------------------------------------------------------------------------
// Host orchestration
// ---------------------------------------------------------------------------
static inline dim3 gemm_grid(int M, int N) { return dim3((N + 127) / 128, (M + 127) / 128); }

extern "C" void kernel_launch(void* const* d_in, const int* in_sizes, int n_in,
                              void* d_out, int out_size, void* d_ws, size_t ws_size,
                              hipStream_t stream) {
  (void)in_sizes; (void)n_in; (void)out_size; (void)ws_size;
  const float* x       = (const float*)d_in[0];
  const float* ln1_g   = (const float*)d_in[4];
  const float* ln1_b   = (const float*)d_in[5];
  const float* lnt_g   = (const float*)d_in[6];
  const float* lnt_b   = (const float*)d_in[7];
  const float* ln2_g   = (const float*)d_in[8];
  const float* ln2_b   = (const float*)d_in[9];
  const float* lncls_g = (const float*)d_in[10];
  const float* lncls_b = (const float*)d_in[11];
  const float* Wqkv_s  = (const float*)d_in[12];
  const float* Wproj_s = (const float*)d_in[13];
  const float* bproj_s = (const float*)d_in[14];
  const float* Wqkv4   = (const float*)d_in[15];
  const float* Wqkv8   = (const float*)d_in[16];
  const float* Wqkv16  = (const float*)d_in[17];
  const float* Wp4     = (const float*)d_in[18];
  const float* bp4     = (const float*)d_in[19];
  const float* Wp8     = (const float*)d_in[20];
  const float* bp8     = (const float*)d_in[21];
  const float* Wp16    = (const float*)d_in[22];
  const float* bp16    = (const float*)d_in[23];
  const float* Wtfc    = (const float*)d_in[24];
  const float* btfc    = (const float*)d_in[25];
  const float* Wfc1    = (const float*)d_in[26];
  const float* bfc1    = (const float*)d_in[27];
  const float* Wfc2    = (const float*)d_in[28];
  const float* bfc2    = (const float*)d_in[29];
  float* outp = (float*)d_out;

  constexpr size_t M_T = 12544;   // B*K*T
  constexpr size_t M8  = 6272;    // B*K*8
  constexpr size_t M4  = 3136;    // B*K*4
  constexpr size_t M_S = 12608;   // B*T*197
  constexpr size_t M_A = 12548;   // B*(1+K*T)
  const float scale = 0.125f;     // 64^-0.5

  // ---- bump allocator over d_ws ----
  char* base = (char*)d_ws;
  size_t off = 0;
  auto alloc = [&](size_t bytes) -> char* {
    char* p = base + off;
    off = (off + bytes + 255) & ~(size_t)255;
    return p;
  };
  float* xt_res   = (float*)alloc(M_T * 768 * 4);
  bf16*  xtn      = (bf16*)alloc(M_T * 768 * 2);
  bf16*  A8       = (bf16*)alloc(M8 * 768 * 2);
  bf16*  A4       = (bf16*)alloc(M4 * 768 * 2);
  bf16*  wqs_t    = (bf16*)alloc((size_t)2304 * 768 * 2);   // transposed weights [N][K]
  bf16*  wq4_t    = (bf16*)alloc((size_t)2304 * 768 * 2);
  bf16*  wq8_t    = (bf16*)alloc((size_t)2304 * 768 * 2);
  bf16*  wq16_t   = (bf16*)alloc((size_t)2304 * 768 * 2);
  bf16*  wproj_t  = (bf16*)alloc((size_t)768 * 768 * 2);
  bf16*  wp4_t    = (bf16*)alloc((size_t)768 * 768 * 2);
  bf16*  wp8_t    = (bf16*)alloc((size_t)768 * 768 * 2);
  bf16*  wp16_t   = (bf16*)alloc((size_t)768 * 768 * 2);
  bf16*  wtfc_t   = (bf16*)alloc((size_t)768 * 768 * 2);
  bf16*  wfc1_t   = (bf16*)alloc((size_t)3072 * 768 * 2);
  bf16*  wfc2_t   = (bf16*)alloc((size_t)768 * 3072 * 2);
  bf16*  qkv16    = (bf16*)alloc(M_T * 2304 * 2);
  bf16*  qkv8     = (bf16*)alloc(M8 * 2304 * 2);
  bf16*  qkv4     = (bf16*)alloc(M4 * 2304 * 2);
  bf16*  o16      = (bf16*)alloc(M_T * 768 * 2);
  bf16*  o8       = (bf16*)alloc(M8 * 768 * 2);
  bf16*  o4       = (bf16*)alloc(M4 * 768 * 2);
  float* x16f     = (float*)alloc(M_T * 768 * 4);
  float* x8f      = (float*)alloc(M8 * 768 * 4);
  float* x4f      = (float*)alloc(M4 * 768 * 4);
  bf16*  res_temp = (bf16*)alloc(M_T * 768 * 2);
  float* xt_full  = (float*)alloc(M_T * 768 * 4);
  bf16*  xs_n     = (bf16*)alloc(M_S * 768 * 2);
  bf16*  qkv_s    = (bf16*)alloc(M_S * 2304 * 2);
  bf16*  out_s    = (bf16*)alloc(M_S * 768 * 2);
  float* res_sp   = (float*)alloc(M_S * 768 * 4);
  float* cls_sp   = (float*)alloc((size_t)64 * 768 * 4);
  float* cls_out  = (float*)alloc((size_t)4 * 768 * 4);
  float* x_cat    = (float*)alloc(M_A * 768 * 4);
  bf16*  h_n      = (bf16*)alloc(M_A * 768 * 2);
  bf16*  hidden   = (bf16*)alloc(M_A * 3072 * 2);

  auto tcvt = [&](const float* src, bf16* dstT, int K, int N) {
    dim3 g((N + 31) / 32, (K + 31) / 32);
    transpose_cvt_kernel<<<g, dim3(32, 8), 0, stream>>>(src, dstT, K, N);
  };
  auto ew_blocks = [](size_t n) { return (int)((n + 255) / 256); };

  // 0) weight transpose + fp32 -> bf16 (W[K][N] -> Wt[N][K])
  tcvt(Wqkv_s,  wqs_t,  768, 2304);
  tcvt(Wqkv4,   wq4_t,  768, 2304);
  tcvt(Wqkv8,   wq8_t,  768, 2304);
  tcvt(Wqkv16,  wq16_t, 768, 2304);
  tcvt(Wproj_s, wproj_t, 768, 768);
  tcvt(Wp4,     wp4_t,  768, 768);
  tcvt(Wp8,     wp8_t,  768, 768);
  tcvt(Wp16,    wp16_t, 768, 768);
  tcvt(Wtfc,    wtfc_t, 768, 768);
  tcvt(Wfc1,    wfc1_t, 768, 3072);
  tcvt(Wfc2,    wfc2_t, 3072, 768);

  // 1) strip cls, temporal LN, window gathers
  strip_cls_kernel<<<ew_blocks(M_T * 768), 256, 0, stream>>>(x, xt_res);
  ln_rows_kernel<<<(int)M_T, 256, 0, stream>>>(xt_res, 768, lnt_g, lnt_b, nullptr, xtn,
                                               (int)M_T);
  gather_win_kernel<<<ew_blocks(M8 * 768), 256, 0, stream>>>(xtn, A8, 8);
  gather_win_kernel<<<ew_blocks(M4 * 768), 256, 0, stream>>>(xtn, A4, 4);

  // 2) temporal QKV GEMMs (bf16 out)
  gemm_bf16_wmma<false, false, false, true><<<gemm_grid((int)M_T, 2304), 256, 0, stream>>>(
      xtn, wq16_t, nullptr, nullptr, nullptr, qkv16, (int)M_T, 2304, 768);
  gemm_bf16_wmma<false, false, false, true><<<gemm_grid((int)M8, 2304), 256, 0, stream>>>(
      A8, wq8_t, nullptr, nullptr, nullptr, qkv8, (int)M8, 2304, 768);
  gemm_bf16_wmma<false, false, false, true><<<gemm_grid((int)M4, 2304), 256, 0, stream>>>(
      A4, wq4_t, nullptr, nullptr, nullptr, qkv4, (int)M4, 2304, 768);

  // 3) temporal attention (L = 16 / 8 / 4)
  temporal_attn_kernel<<<dim3(784, 12), 128, 0, stream>>>(qkv16, o16, 16, scale);
  temporal_attn_kernel<<<dim3(784, 12), 128, 0, stream>>>(qkv8, o8, 8, scale);
  temporal_attn_kernel<<<dim3(784, 12), 128, 0, stream>>>(qkv4, o4, 4, scale);

  // 4) temporal output projections (fp32 out + bias)
  gemm_bf16_wmma<true, false, false, false><<<gemm_grid((int)M_T, 768), 256, 0, stream>>>(
      o16, wp16_t, bp16, nullptr, x16f, nullptr, (int)M_T, 768, 768);
  gemm_bf16_wmma<true, false, false, false><<<gemm_grid((int)M8, 768), 256, 0, stream>>>(
      o8, wp8_t, bp8, nullptr, x8f, nullptr, (int)M8, 768, 768);
  gemm_bf16_wmma<true, false, false, false><<<gemm_grid((int)M4, 768), 256, 0, stream>>>(
      o4, wp4_t, bp4, nullptr, x4f, nullptr, (int)M4, 768, 768);

  // 5) hierarchical merge -> res_temp (bf16); xt_full = res_temp@Wtfc + btfc + xt_res
  merge_temporal_kernel<<<ew_blocks(M_T * 768), 256, 0, stream>>>(x16f, x8f, x4f, res_temp);
  gemm_bf16_wmma<true, true, false, false><<<gemm_grid((int)M_T, 768), 256, 0, stream>>>(
      res_temp, wtfc_t, btfc, xt_res, xt_full, nullptr, (int)M_T, 768, 768);

  // 6) spatial path: gather+LN -> QKV -> WMMA attention -> projection
  ln_xs_kernel<<<(int)M_S, 256, 0, stream>>>(x, xt_full, ln1_g, ln1_b, xs_n);
  gemm_bf16_wmma<false, false, false, true><<<gemm_grid((int)M_S, 2304), 256, 0, stream>>>(
      xs_n, wqs_t, nullptr, nullptr, nullptr, qkv_s, (int)M_S, 2304, 768);
  spatial_attn_kernel<<<dim3(13, 12, 64), 128, 0, stream>>>(qkv_s, out_s, scale);
  gemm_bf16_wmma<true, false, false, false><<<gemm_grid((int)M_S, 768), 256, 0, stream>>>(
      out_s, wproj_t, bproj_s, nullptr, res_sp, nullptr, (int)M_S, 768, 768);

  // 7) CLS aggregation
  ln_rows_kernel<<<64, 256, 0, stream>>>(res_sp, (size_t)197 * 768, lncls_g, lncls_b,
                                         cls_sp, nullptr, 64);
  cls_attn_kernel<<<4, 256, 0, stream>>>(cls_sp, cls_out);

  // 8) concat + residual, LN, MLP
  build_xcat_kernel<<<ew_blocks(M_A * 768), 256, 0, stream>>>(x, xt_full, res_sp, cls_out,
                                                              x_cat);
  ln_rows_kernel<<<(int)M_A, 256, 0, stream>>>(x_cat, 768, ln2_g, ln2_b, nullptr, h_n,
                                               (int)M_A);
  gemm_bf16_wmma<true, false, true, true><<<gemm_grid((int)M_A, 3072), 256, 0, stream>>>(
      h_n, wfc1_t, bfc1, nullptr, nullptr, hidden, (int)M_A, 3072, 768);
  gemm_bf16_wmma<true, true, false, false><<<gemm_grid((int)M_A, 768), 256, 0, stream>>>(
      hidden, wfc2_t, bfc2, x_cat, outp, nullptr, (int)M_A, 768, 3072);
}
